// NoisyTopkRouter_63419487093415
// MI455X (gfx1250) — compile-verified, matrix operand order checked by reading, not run on hardware
//
#include <hip/hip_runtime.h>

// ---------------------------------------------------------------------------
// NoisyTopkRouter for MI455X (gfx1250, wave32, WMMA + async-to-LDS staging)
//   x:[8,8192,1024] f32, W_route/W_noise:[1024,64] f32, b_*:[64], noise:[B,T,64]
//   out = router_output [B,T,64] f32  ++  indices [B,T,2] i32
// ---------------------------------------------------------------------------

#define D_DIM   1024
#define E_DIM   64
#define TOK_PER_BLOCK 128
#define THREADS 256
#define BT_TOTAL 65536

#define FRAG_BYTES   262144   // 256 KB bf16 weight fragments (both matrices)
#define STAGE_BYTES   32768   // 32 KB fp32 async staging chunk

typedef __attribute__((ext_vector_type(16))) __bf16 v16bf;
typedef __attribute__((ext_vector_type(8)))  float  v8f;

union BF16x16 { v16bf v; __bf16 e[16]; };
union F32x8   { v8f   v; float  e[8];  };

__global__ __launch_bounds__(THREADS)
void noisy_topk_router_kernel(const float* __restrict__ x,
                              const float* __restrict__ W_route,
                              const float* __restrict__ b_route,
                              const float* __restrict__ W_noise,
                              const float* __restrict__ b_noise,
                              const float* __restrict__ noise,
                              float* __restrict__ out,
                              int*   __restrict__ idx_out)
{
    // [0, 256K): bf16 WMMA B-fragments, both matrices (reused for logits later)
    // [256K, 288K): fp32 staging buffer for async global->LDS weight copies
    __shared__ __align__(1024) unsigned char smem[FRAG_BYTES + STAGE_BYTES];
    unsigned char* stage = smem + FRAG_BYTES;

    const int tid  = threadIdx.x;
    const int lane = tid & 31;
    const int wave = tid >> 5;
    const int blockTok = blockIdx.x * TOK_PER_BLOCK;

    // ---- stage both weight matrices: async global->LDS, then swizzle ----
    // 16 chunks x 32 KB covers W_route (256 KB fp32) then W_noise (256 KB).
    // Fragment layout (ISA 7.12.2, 16-bit 32x16 operand): lanes 0-15 hold
    // K{0..7,16..23}, lanes 16-31 hold K{8..15,24..31}; fragment = 1 KB.
    const unsigned stage_lds = (unsigned)(uintptr_t)stage;  // LDS byte address
    for (int c = 0; c < 16; ++c) {
        const float* wsrc = (c < 8) ? (W_route + c * 8192)
                                    : (W_noise + (c - 8) * 8192);
        // 2048 x b128 transfers, 8 per thread, ASYNCcnt-tracked
#pragma unroll
        for (int t = 0; t < 8; ++t) {
            const int elt = t * THREADS + tid;            // 0..2047
            const unsigned ldsa = stage_lds + elt * 16;
            const float*   ga   = wsrc + elt * 4;
            asm volatile("global_load_async_to_lds_b128 %0, %1, off"
                         :: "v"(ldsa), "v"(ga) : "memory");
        }
        asm volatile("s_wait_asynccnt 0x0" ::: "memory");
        __syncthreads();   // all waves' async transfers landed

        const float* sf = (const float*)stage;
#pragma unroll
        for (int t = 0; t < 32; ++t) {
            const int j = t * THREADS + tid;              // 0..8191 floats
            const int f = c * 8192 + j;                   // flat over both mats
            const int mat = f >> 16;
            const int r   = f & 0xFFFF;
            const int k   = r >> 6;                       // K row of W
            const int e   = r & 63;                       // expert column
            const float w = sf[j];
            const int kc = k >> 5, nt = e >> 4, kk = k & 31, ln = e & 15;
            const int q  = kk >> 3;
            const int sl = ln + ((q & 1) << 4);
            const int p  = kk - 8 * ((q + 1) >> 1);
            const int off = (((mat << 7) + (kc << 2) + nt) << 10) + sl * 32 + p * 2;
            *(__bf16*)(smem + off) = (__bf16)w;
        }
        __syncthreads();   // before next chunk overwrites staging
    }

    // ---- K loop: 16 tokens per wave, 4 N-tiles x 2 matrices accumulators ----
    v8f accR[4] = {};
    v8f accN[4] = {};

    const int rowTok = blockTok + wave * 16 + (lane & 15);
    const float* __restrict__ xr = x + (size_t)rowTok * D_DIM;
    const int kshift = (lane >> 4) * 8;   // lanes 16-31 take the +8 K groups

    for (int k0 = 0; k0 < D_DIM; k0 += 32) {
        const float* xp = xr + k0 + kshift;
        const float4 c0 = *(const float4*)(xp + 0);
        const float4 c1 = *(const float4*)(xp + 4);
        const float4 c2 = *(const float4*)(xp + 16);
        const float4 c3 = *(const float4*)(xp + 20);
        if (k0 + 32 < D_DIM) __builtin_prefetch(xp + 32, 0, 3);

        BF16x16 a;
        const float t[16] = { c0.x, c0.y, c0.z, c0.w, c1.x, c1.y, c1.z, c1.w,
                              c2.x, c2.y, c2.z, c2.w, c3.x, c3.y, c3.z, c3.w };
#pragma unroll
        for (int j = 0; j < 16; ++j) a.e[j] = (__bf16)t[j];

        // fetch all 8 B fragments first (one dscnt wait covers the clause),
        // then issue the 8 WMMAs
        const int kc4 = (k0 >> 5) << 2;
        v16bf bfr[8];
#pragma unroll
        for (int nt = 0; nt < 4; ++nt) {
            bfr[nt]     = *(const v16bf*)(smem + ((kc4 + nt)       << 10) + lane * 32);
            bfr[4 + nt] = *(const v16bf*)(smem + ((128 + kc4 + nt) << 10) + lane * 32);
        }
#pragma unroll
        for (int nt = 0; nt < 4; ++nt) {
            accR[nt] = __builtin_amdgcn_wmma_f32_16x16x32_bf16(
                false, a.v, false, bfr[nt], (short)0, accR[nt], false, false);
            accN[nt] = __builtin_amdgcn_wmma_f32_16x16x32_bf16(
                false, a.v, false, bfr[4 + nt], (short)0, accN[nt], false, false);
        }
    }

    // all waves done with the weight fragments -> reuse LDS for logits
    __syncthreads();
    float* nl = (float*)smem;   // [128 tokens][64 experts]

    // C/D fragment layout: VGPR g, lane L -> M = g + 8*(L/16), N = L%16
    const int halfsel = lane >> 4;
    const int col     = lane & 15;
#pragma unroll
    for (int nt = 0; nt < 4; ++nt) {
        const int e = nt * 16 + col;
        const float brt = b_route[e];
        const float bns = b_noise[e];
        F32x8 rr, nn;
        rr.v = accR[nt];
        nn.v = accN[nt];
#pragma unroll
        for (int g = 0; g < 8; ++g) {
            const int m     = g + 8 * halfsel;
            const int token = blockTok + wave * 16 + m;
            const float lg  = rr.e[g] + brt;
            const float nz  = nn.e[g] + bns;
            const float sp  = (nz > 20.f) ? nz : log1pf(__expf(nz)); // softplus
            const float nv  = noise[(size_t)token * E_DIM + e];
            nl[(wave * 16 + m) * E_DIM + e] = lg + nv * sp;
        }
    }
    __syncthreads();

    // ---- zero-fill output rows owned by this wave (vectorized) ----
    float* orow = out + (size_t)(blockTok + wave * 16) * E_DIM;
    const float4 z4 = make_float4(0.f, 0.f, 0.f, 0.f);
#pragma unroll
    for (int i = 0; i < 8; ++i)
        ((float4*)orow)[lane + i * 32] = z4;

    // ---- top-2 + softmax scatter: one token per lane (lanes 0-15) ----
    if (lane < 16) {
        const float* row = nl + (wave * 16 + lane) * E_DIM;
        float best = -3.402823466e+38f; int bi = 0;
        for (int e = 0; e < E_DIM; ++e) {
            const float v = row[e];
            if (v > best) { best = v; bi = e; }
        }
        float sec = -3.402823466e+38f; int si = 0;
        for (int e = 0; e < E_DIM; ++e) {
            const float v = row[e];
            if (e != bi && v > sec) { sec = v; si = e; }
        }
        const float p1 = __expf(sec - best);
        const float dn = 1.f + p1;
        const int token = blockTok + wave * 16 + lane;
        out[(size_t)token * E_DIM + bi] = 1.f / dn;   // ordered after zero stores
        out[(size_t)token * E_DIM + si] = p1 / dn;
        idx_out[token * 2 + 0] = bi;
        idx_out[token * 2 + 1] = si;
    }
}

extern "C" void kernel_launch(void* const* d_in, const int* in_sizes, int n_in,
                              void* d_out, int out_size, void* d_ws, size_t ws_size,
                              hipStream_t stream) {
    const float* x       = (const float*)d_in[0];
    const float* W_route = (const float*)d_in[1];
    const float* b_route = (const float*)d_in[2];
    const float* W_noise = (const float*)d_in[3];
    const float* b_noise = (const float*)d_in[4];
    const float* noise   = (const float*)d_in[5];

    float* out = (float*)d_out;
    int*   idx = (int*)((float*)d_out + (size_t)BT_TOTAL * E_DIM);

    dim3 grid(BT_TOTAL / TOK_PER_BLOCK);   // 512 blocks
    dim3 block(THREADS);                   // 8 waves of 32
    noisy_topk_router_kernel<<<grid, block, 0, stream>>>(
        x, W_route, b_route, W_noise, b_noise, noise, out, idx);
}
